// _SlotAttention_50122268344786
// MI455X (gfx1250) — matrix operand order repaired and checked by reading, not compiled, not run on hardware
//
#include <hip/hip_runtime.h>
#include <hip/hip_bf16.h>

typedef __attribute__((ext_vector_type(16))) _Float16 v16h;
typedef __attribute__((ext_vector_type(8)))  _Float16 v8h;
typedef __attribute__((ext_vector_type(8)))  float    v8f;

#define LN_EPS 1e-5f

// ---------------------------------------------------------------------------
// WMMA helpers: D = A(16x32 f16) x B(32x16 f16) + C(16x16 f32)
// A layout: lane -> M=lane&15; elems 0-7 = K[kb..kb+7], elems 8-15 =
//           K[kb+16..kb+23], kb = (lane>>4)*8  (two 16B runs).
// B layout (N-major storage Bt[N][K]): lane -> N=lane&15; same K order as A
//           (two 16B runs, kb = (lane>>4)*8).
// C/D: element i -> M = i + 8*(lane>>4), N = lane&15.
// ---------------------------------------------------------------------------
__device__ __forceinline__ v8f wmma_f16(v16h a, v16h b, v8f c) {
  return __builtin_amdgcn_wmma_f32_16x16x32_f16(false, a, false, b, (short)0, c,
                                                false, false);
}

__device__ __forceinline__ v16h load_frag_a(const _Float16* p) {
  v8h lo = *(const v8h*)p;
  v8h hi = *(const v8h*)(p + 16);
  v16h r;
#pragma unroll
  for (int i = 0; i < 8; ++i) { r[i] = lo[i]; r[i + 8] = hi[i]; }
  return r;
}

__device__ __forceinline__ v16h load_frag_b(const _Float16* p) {
  return load_frag_a(p);
}

// ---------------------------------------------------------------------------
// Weight prep: in f32 [K][N] row-major -> out f16 [N][K] (N-major for B frags)
// ---------------------------------------------------------------------------
__global__ void transpose_f32_to_f16_kernel(const float* __restrict__ in,
                                            _Float16* __restrict__ out,
                                            int K, int N) {
  long i = (long)blockIdx.x * blockDim.x + threadIdx.x;
  long total = (long)K * N;
  if (i >= total) return;
  int n = (int)(i / K), k = (int)(i % K);
  out[i] = (_Float16)in[(long)k * N + n];
}

__global__ void zero_f16_kernel(_Float16* __restrict__ p, long n) {
  long i = (long)blockIdx.x * blockDim.x + threadIdx.x;
  if (i < n) p[i] = (_Float16)0.f;
}

// ---------------------------------------------------------------------------
// slots = slot_mu + clip(softplus(slot_sigma), 0.1, 2.0) * noise
// ---------------------------------------------------------------------------
__global__ void slots_init_kernel(const float* __restrict__ noise,
                                  const float* __restrict__ mu,
                                  const float* __restrict__ sigma,
                                  float* __restrict__ slots, int total) {
  int i = blockIdx.x * blockDim.x + threadIdx.x;
  if (i >= total) return;
  int d = i & 255;
  float sp = log1pf(__expf(sigma[d]));
  sp = fminf(fmaxf(sp, 0.1f), 2.0f);
  slots[i] = mu[d] + sp * noise[i];
}

// ---------------------------------------------------------------------------
// Fused input LayerNorm + k/v projection.
// Block: 256 threads (8 waves), 64 rows. Phase 1: LN -> f16 in LDS.
// Phase 2: waves 0-3 compute k (m-tiles 0-3), waves 4-7 compute v.
// k stored row-major f16 [token][256] (LDS-staged contiguous stores);
// v stored transposed f16 [b][d][4096] (natively contiguous stores).
// ---------------------------------------------------------------------------
#define LNP_ROWS 64
__global__ void ln_project_kernel(const float* __restrict__ inputs,
                                  const float* __restrict__ gam,
                                  const float* __restrict__ bet,
                                  const _Float16* __restrict__ WkT,
                                  const _Float16* __restrict__ WvT,
                                  _Float16* __restrict__ k16,
                                  _Float16* __restrict__ vt16) {
  __shared__ _Float16 x16[LNP_ROWS][256 + 8];
  __shared__ _Float16 stg[8][16][24];  // per-wave C-tile transpose staging
  const int tid = threadIdx.x;
  const int wave = tid >> 5, lane = tid & 31;
  const long rowBase = (long)blockIdx.x * LNP_ROWS;

  // Phase 1: LayerNorm 8 rows per wave
  for (int rr = 0; rr < 8; ++rr) {
    const int rl = wave * 8 + rr;
    const float* src = inputs + (rowBase + rl) * 256 + lane * 8;
    float4 a = *(const float4*)(src);
    float4 c = *(const float4*)(src + 4);
    float vals[8] = {a.x, a.y, a.z, a.w, c.x, c.y, c.z, c.w};
    float s = 0.f, s2 = 0.f;
#pragma unroll
    for (int j = 0; j < 8; ++j) { s += vals[j]; s2 += vals[j] * vals[j]; }
#pragma unroll
    for (int m = 16; m >= 1; m >>= 1) {
      s += __shfl_xor(s, m, 32);
      s2 += __shfl_xor(s2, m, 32);
    }
    float mean = s * (1.f / 256.f);
    float var = s2 * (1.f / 256.f) - mean * mean;
    float rstd = rsqrtf(var + LN_EPS);
#pragma unroll
    for (int j = 0; j < 8; ++j) {
      int col = lane * 8 + j;
      x16[rl][col] = (_Float16)((vals[j] - mean) * rstd * gam[col] + bet[col]);
    }
  }
  __syncthreads();

  // Phase 2: GEMM out of LDS (A frags hoisted by compiler; B pipelined)
  const int mat = wave >> 2;    // 0 -> k, 1 -> v
  const int mtile = wave & 3;
  const _Float16* WT = mat ? WvT : WkT;
  const int r = lane & 15, kg = lane >> 4;

  for (int ntile = 0; ntile < 16; ++ntile) {
    const _Float16* Bb = WT + (long)(ntile * 16 + r) * 256 + kg * 8;
    v8f acc = {};
    v16h bcur = load_frag_b(Bb);
#pragma unroll
    for (int kk = 0; kk < 8; ++kk) {
      v16h bnxt = {};
      if (kk < 7) bnxt = load_frag_b(Bb + (kk + 1) * 32);
      v16h af = load_frag_a(&x16[mtile * 16 + r][kk * 32 + kg * 8]);
      acc = wmma_f16(af, bcur, acc);
      bcur = bnxt;
    }
    const int col = ntile * 16 + r;
    if (mat == 0) {
      // stage C tile in LDS, then contiguous row stores (same-wave DS order)
#pragma unroll
      for (int i = 0; i < 8; ++i)
        stg[wave][i + 8 * kg][r] = (_Float16)acc[i];
      if (lane < 16) {
        const _Float16* sp = &stg[wave][lane][0];
        v8h lo = *(const v8h*)sp;
        v8h hi = *(const v8h*)(sp + 8);
        _Float16* dst = k16 + (rowBase + mtile * 16 + lane) * 256 + ntile * 16;
        *(v8h*)dst = lo;
        *(v8h*)(dst + 8) = hi;
      }
    } else {
      const long token0 = rowBase + mtile * 16 + 8 * kg;
      const long b = token0 >> 12;
      const long tin = token0 & 4095;
      _Float16* dst = vt16 + (b * 256 + col) * 4096 + tin;
      v8h h;
#pragma unroll
      for (int i = 0; i < 8; ++i) h[i] = (_Float16)acc[i];
      *(v8h*)dst = h;
    }
  }
}

// ---------------------------------------------------------------------------
// Generic LN over rows of 256 (slots-side). One wave per row, 8 waves/block.
// ---------------------------------------------------------------------------
__global__ void ln256_kernel(const float* __restrict__ in,
                             const float* __restrict__ gam,
                             const float* __restrict__ bet,
                             _Float16* __restrict__ out16,
                             _Float16* __restrict__ raw16) {
  const int wave = threadIdx.x >> 5, lane = threadIdx.x & 31;
  const long row = (long)blockIdx.x * 8 + wave;
  const float* src = in + row * 256 + lane * 8;
  float4 a = *(const float4*)(src);
  float4 c = *(const float4*)(src + 4);
  float vals[8] = {a.x, a.y, a.z, a.w, c.x, c.y, c.z, c.w};
  float s = 0.f, s2 = 0.f;
#pragma unroll
  for (int j = 0; j < 8; ++j) { s += vals[j]; s2 += vals[j] * vals[j]; }
#pragma unroll
  for (int m = 16; m >= 1; m >>= 1) {
    s += __shfl_xor(s, m, 32);
    s2 += __shfl_xor(s2, m, 32);
  }
  float mean = s * (1.f / 256.f);
  float var = s2 * (1.f / 256.f) - mean * mean;
  float rstd = rsqrtf(var + LN_EPS);
#pragma unroll
  for (int j = 0; j < 8; ++j) {
    int col = lane * 8 + j;
    out16[row * 256 + col] = (_Float16)((vals[j] - mean) * rstd * gam[col] + bet[col]);
    if (raw16) raw16[row * 256 + col] = (_Float16)vals[j];
  }
}

// ---------------------------------------------------------------------------
// dots_t[b][s][n] = scale * sum_d q[b][s][d] * k[b][n][d]
// A = q (loop-invariant, preloaded, s>=8 zero-padded), B = row-major k16.
// grid (8, 32), block 256: wave handles 4 n-tiles (of 256 per batch).
// ---------------------------------------------------------------------------
__global__ void dots_kernel(const _Float16* __restrict__ k16,
                            const _Float16* __restrict__ q16,
                            float* __restrict__ dots_t, float scale) {
  const int b = blockIdx.y;
  const int wave = threadIdx.x >> 5, lane = threadIdx.x & 31;
  const int r = lane & 15, kg = lane >> 4;

  v16h aq[8];
  const _Float16* Ab = q16 + ((long)b * 8 + r) * 256;
#pragma unroll
  for (int kk = 0; kk < 8; ++kk) {
    v16h t = {};
    if (r < 8) t = load_frag_a(Ab + kk * 32 + kg * 8);
    aq[kk] = t;
  }

#pragma unroll
  for (int j = 0; j < 4; ++j) {
    const int nt = blockIdx.x * 8 + wave + 64 * j;
    const _Float16* Bb = k16 + ((long)b * 4096 + nt * 16 + r) * 256 + kg * 8;
    v8f acc = {};
    v16h bcur = load_frag_b(Bb);
#pragma unroll
    for (int kk = 0; kk < 8; ++kk) {
      v16h bnxt = {};
      if (kk < 7) bnxt = load_frag_b(Bb + (kk + 1) * 32);
      acc = wmma_f16(aq[kk], bcur, acc);
      bcur = bnxt;
    }
    if (kg == 0) {  // element i -> s = i (valid), n = nt*16 + r
      float* d = dots_t + ((long)b * 8) * 4096 + nt * 16 + r;
#pragma unroll
      for (int i = 0; i < 8; ++i) d[(long)i * 4096] = acc[i] * scale;
    }
  }
}

// ---------------------------------------------------------------------------
// Softmax over n (4096) per (b,s); coalesced reads of dots_t[b][s][:].
// Writes attn_t[b][16][4096] f16 (rows 8..15 stay zero).
// ---------------------------------------------------------------------------
__global__ void softmax_kernel(const float* __restrict__ dots_t,
                               _Float16* __restrict__ attn_t) {
  const int s = blockIdx.x, b = blockIdx.y;
  const int tid = threadIdx.x;  // 256 threads
  const float* src = dots_t + ((long)b * 8 + s) * 4096;
  float loc[16];
  float mx = -3.0e38f;
#pragma unroll
  for (int i = 0; i < 16; ++i) {
    loc[i] = src[i * 256 + tid];
    mx = fmaxf(mx, loc[i]);
  }
  __shared__ float red[256];
  red[tid] = mx;
  __syncthreads();
  for (int st = 128; st > 0; st >>= 1) {
    if (tid < st) red[tid] = fmaxf(red[tid], red[tid + st]);
    __syncthreads();
  }
  mx = red[0];
  __syncthreads();
  float sum = 0.f;
#pragma unroll
  for (int i = 0; i < 16; ++i) {
    loc[i] = __expf(loc[i] - mx);
    sum += loc[i];
  }
  red[tid] = sum;
  __syncthreads();
  for (int st = 128; st > 0; st >>= 1) {
    if (tid < st) red[tid] += red[tid + st];
    __syncthreads();
  }
  const float inv = 1.0f / red[0];
  _Float16* dst = attn_t + ((long)b * 16 + s) * 4096;
#pragma unroll
  for (int i = 0; i < 16; ++i)
    dst[i * 256 + tid] = (_Float16)(loc[i] * inv);
}

// ---------------------------------------------------------------------------
// updates[b][s][d] = sum_n attn_t[b][s][n] * v_t[b][d][n]   (K = 4096)
// 8 waves/block: K split across waves (512 each, 16 pipelined WMMA steps),
// then LDS cross-wave reduction. grid (16, 32), block 256.
// ---------------------------------------------------------------------------
__global__ void updates_kernel(const _Float16* __restrict__ attn_t,
                               const _Float16* __restrict__ vt16,
                               _Float16* __restrict__ updates16) {
  __shared__ float part[8][16][17];
  const int ntile = blockIdx.x, b = blockIdx.y;
  const int wave = threadIdx.x >> 5, lane = threadIdx.x & 31;
  const int r = lane & 15, kg = lane >> 4;
  const _Float16* Ab = attn_t + ((long)b * 16 + r) * 4096 + wave * 512 + kg * 8;
  const _Float16* Bb =
      vt16 + ((long)b * 256 + ntile * 16 + r) * 4096 + wave * 512 + kg * 8;

  v8f acc = {};
  v16h acur = load_frag_a(Ab);
  v16h bcur = load_frag_b(Bb);
#pragma unroll
  for (int kk = 0; kk < 16; ++kk) {
    v16h anxt = {}, bnxt = {};
    if (kk < 15) {
      anxt = load_frag_a(Ab + (kk + 1) * 32);
      bnxt = load_frag_b(Bb + (kk + 1) * 32);
    }
    acc = wmma_f16(acur, bcur, acc);
    acur = anxt;
    bcur = bnxt;
  }
#pragma unroll
  for (int i = 0; i < 8; ++i) part[wave][i + 8 * kg][r] = acc[i];
  __syncthreads();

  const int s = threadIdx.x >> 4, col = threadIdx.x & 15;
  if (s < 8) {
    float sum = 0.f;
#pragma unroll
    for (int w = 0; w < 8; ++w) sum += part[w][s][col];
    updates16[((long)b * 8 + s) * 256 + ntile * 16 + col] = (_Float16)sum;
  }
}

// ---------------------------------------------------------------------------
// Generic WMMA GEMM: C[M][N] = A16[M][K] @ Bt16[N][K]^T (+ bias[N])
// grid = (N/16, M/16), block = 32. Epilogue modes:
//  0: f32 store   1: f16 store   2: relu -> f16 store   3: f32 accumulate (+=)
// ---------------------------------------------------------------------------
__global__ void gemm16_kernel(const _Float16* __restrict__ A,
                              const _Float16* __restrict__ Bt,
                              const float* __restrict__ bias,
                              void* __restrict__ C, int K, int N, int mode) {
  const int lane = threadIdx.x & 31;
  const int r = lane & 15, kg = lane >> 4;
  const _Float16* Ab = A + ((long)blockIdx.y * 16 + r) * K + kg * 8;
  const _Float16* Bb = Bt + ((long)blockIdx.x * 16 + r) * K + kg * 8;
  v8f acc = {};
  const int ksteps = K >> 5;
  v16h acur = load_frag_a(Ab);
  v16h bcur = load_frag_b(Bb);
  for (int kk = 0; kk < ksteps; ++kk) {
    v16h anxt = {}, bnxt = {};
    if (kk + 1 < ksteps) {
      anxt = load_frag_a(Ab + (kk + 1) * 32);
      bnxt = load_frag_b(Bb + (kk + 1) * 32);
    }
    acc = wmma_f16(acur, bcur, acc);
    acur = anxt;
    bcur = bnxt;
  }
  const int col = blockIdx.x * 16 + r;
  const float bv = bias ? bias[col] : 0.f;
  const long row0 = (long)blockIdx.y * 16 + 8 * kg;
  if (mode == 0) {
    float* Cf = (float*)C;
#pragma unroll
    for (int i = 0; i < 8; ++i) Cf[(row0 + i) * N + col] = acc[i] + bv;
  } else if (mode == 1) {
    _Float16* Ch = (_Float16*)C;
#pragma unroll
    for (int i = 0; i < 8; ++i) Ch[(row0 + i) * N + col] = (_Float16)(acc[i] + bv);
  } else if (mode == 2) {
    _Float16* Ch = (_Float16*)C;
#pragma unroll
    for (int i = 0; i < 8; ++i)
      Ch[(row0 + i) * N + col] = (_Float16)fmaxf(acc[i] + bv, 0.f);
  } else {
    float* Cf = (float*)C;
#pragma unroll
    for (int i = 0; i < 8; ++i) Cf[(row0 + i) * N + col] += acc[i] + bv;
  }
}

// ---------------------------------------------------------------------------
// GRU gating (torch GRUCell, gate order r,z,n):
// slots = (1-z)*tanh(xn + r*hn) + z*slots_prev
// ---------------------------------------------------------------------------
__global__ void gru_gate_kernel(const float* __restrict__ gx,
                                const float* __restrict__ gh,
                                float* __restrict__ slots) {
  const int i = blockIdx.x * blockDim.x + threadIdx.x;
  if (i >= 65536) return;
  const int row = i >> 8, col = i & 255;
  const float* gxr = gx + (long)row * 768;
  const float* ghr = gh + (long)row * 768;
  float r = 1.f / (1.f + __expf(-(gxr[col] + ghr[col])));
  float z = 1.f / (1.f + __expf(-(gxr[256 + col] + ghr[256 + col])));
  float ng = tanhf(gxr[512 + col] + r * ghr[512 + col]);
  float prev = slots[i];
  slots[i] = (1.f - z) * ng + z * prev;
}

// ---------------------------------------------------------------------------
// Host orchestration
// ---------------------------------------------------------------------------
extern "C" void kernel_launch(void* const* d_in, const int* in_sizes, int n_in,
                              void* d_out, int out_size, void* d_ws, size_t ws_size,
                              hipStream_t stream) {
  const float* inputs   = (const float*)d_in[0];
  const float* noise    = (const float*)d_in[1];
  const float* ln_in_g  = (const float*)d_in[2];
  const float* ln_in_b  = (const float*)d_in[3];
  const float* ln_sl_g  = (const float*)d_in[4];
  const float* ln_sl_b  = (const float*)d_in[5];
  const float* ln_ml_g  = (const float*)d_in[6];
  const float* ln_ml_b  = (const float*)d_in[7];
  const float* slot_mu  = (const float*)d_in[8];
  const float* slot_sig = (const float*)d_in[9];
  const float* Wq = (const float*)d_in[10];
  const float* Wk = (const float*)d_in[11];
  const float* Wv = (const float*)d_in[12];
  const float* W_ih = (const float*)d_in[13];
  const float* W_hh = (const float*)d_in[14];
  const float* b_ih = (const float*)d_in[15];
  const float* b_hh = (const float*)d_in[16];
  const float* W1 = (const float*)d_in[17];
  const float* b1 = (const float*)d_in[18];
  const float* W2 = (const float*)d_in[19];
  const float* b2 = (const float*)d_in[20];

  char* p = (char*)d_ws;
  auto carve = [&](size_t bytes) -> void* {
    void* r = (void*)p;
    p += (bytes + 255) & ~(size_t)255;
    return r;
  };
  const long BT = 32L * 4096;            // tokens
  _Float16* k16   = (_Float16*)carve(BT * 256 * 2);
  _Float16* vt16  = (_Float16*)carve(BT * 256 * 2);
  _Float16* WkT   = (_Float16*)carve(256 * 256 * 2);
  _Float16* WvT   = (_Float16*)carve(256 * 256 * 2);
  _Float16* WqT   = (_Float16*)carve(256 * 256 * 2);
  _Float16* WihT  = (_Float16*)carve(768 * 256 * 2);
  _Float16* WhhT  = (_Float16*)carve(768 * 256 * 2);
  _Float16* W1T   = (_Float16*)carve(512 * 256 * 2);
  _Float16* W2T   = (_Float16*)carve(256 * 512 * 2);
  float*    slots = (float*)carve(65536 * 4);
  _Float16* s16   = (_Float16*)carve(65536 * 2);
  _Float16* sl16  = (_Float16*)carve(65536 * 2);
  _Float16* m16   = (_Float16*)carve(65536 * 2);
  _Float16* q16   = (_Float16*)carve(65536 * 2);
  _Float16* upd16 = (_Float16*)carve(65536 * 2);
  float*    dotsT = (float*)carve(32L * 8 * 4096 * 4);
  _Float16* attnT = (_Float16*)carve(32L * 16 * 4096 * 2);
  float*    gx    = (float*)carve(256L * 768 * 4);
  float*    gh    = (float*)carve(256L * 768 * 4);
  _Float16* h1    = (_Float16*)carve(256L * 512 * 2);

  // Weight prep (transposed f16)
  auto tr = [&](const float* in, _Float16* out, int K, int N) {
    long tot = (long)K * N;
    transpose_f32_to_f16_kernel<<<dim3((tot + 255) / 256), dim3(256), 0, stream>>>(
        in, out, K, N);
  };
  tr(Wq, WqT, 256, 256);
  tr(Wk, WkT, 256, 256);
  tr(Wv, WvT, 256, 256);
  tr(W_ih, WihT, 256, 768);
  tr(W_hh, WhhT, 256, 768);
  tr(W1, W1T, 256, 512);
  tr(W2, W2T, 512, 256);

  // Zero padded attn rows (rows 8..15 per batch stay zero)
  {
    long n = 32L * 16 * 4096;
    zero_f16_kernel<<<dim3((n + 255) / 256), dim3(256), 0, stream>>>(attnT, n);
  }

  // Slot init
  slots_init_kernel<<<dim3(256), dim3(256), 0, stream>>>(noise, slot_mu, slot_sig,
                                                         slots, 65536);

  // Fused input-LN + k/v projection
  ln_project_kernel<<<dim3(BT / LNP_ROWS), dim3(256), 0, stream>>>(
      inputs, ln_in_g, ln_in_b, WkT, WvT, k16, vt16);

  const float scale = 0.0625f;  // 256^-0.5
  for (int it = 0; it < 3; ++it) {
    // s = LN(slots) -> s16 ; raw slots -> sl16 (for GRU gh)
    ln256_kernel<<<dim3(32), dim3(256), 0, stream>>>(slots, ln_sl_g, ln_sl_b, s16, sl16);
    // q = s @ Wq  (f16 out)
    gemm16_kernel<<<dim3(16, 16), dim3(32), 0, stream>>>(s16, WqT, nullptr, q16,
                                                         256, 256, 1);
    // dots_t = scale * q . k
    dots_kernel<<<dim3(8, 32), dim3(256), 0, stream>>>(k16, q16, dotsT, scale);
    // softmax over n -> attn_t (f16, padded)
    softmax_kernel<<<dim3(8, 32), dim3(256), 0, stream>>>(dotsT, attnT);
    // updates = attn^T @ v
    updates_kernel<<<dim3(16, 32), dim3(256), 0, stream>>>(attnT, vt16, upd16);
    // GRU gates
    gemm16_kernel<<<dim3(48, 16), dim3(32), 0, stream>>>(upd16, WihT, b_ih, gx,
                                                         256, 768, 0);
    gemm16_kernel<<<dim3(48, 16), dim3(32), 0, stream>>>(sl16, WhhT, b_hh, gh,
                                                         256, 768, 0);
    gru_gate_kernel<<<dim3(256), dim3(256), 0, stream>>>(gx, gh, slots);
    // MLP with residual
    ln256_kernel<<<dim3(32), dim3(256), 0, stream>>>(slots, ln_ml_g, ln_ml_b, m16,
                                                     nullptr);
    gemm16_kernel<<<dim3(32, 16), dim3(32), 0, stream>>>(m16, W1T, b1, h1,
                                                         256, 512, 2);
    gemm16_kernel<<<dim3(16, 16), dim3(32), 0, stream>>>(h1, W2T, b2, slots,
                                                         512, 256, 3);
  }

  hipMemcpyAsync(d_out, slots, 65536 * sizeof(float), hipMemcpyDeviceToDevice,
                 stream);
}
